// EGNN_16217796509990
// MI455X (gfx1250) — matrix operand order repaired and checked by reading, not compiled, not run on hardware
//
#include <hip/hip_runtime.h>
#include <stdint.h>

// EGNN forward, fused, bf16 WMMA (gfx1250 / CDNA5, wave32).
// Edge MLP + segment-sum fused so the [640000,256] intermediate never hits HBM.
// Edge kernel: 64-row M-tiles, each wave = 4 row-slabs x 2 col-tiles -> each B
// fragment feeds 4 WMMAs (halves L2 weight traffic vs 32-row tiles).

typedef __attribute__((ext_vector_type(16))) __bf16 bf16x16;
typedef __attribute__((ext_vector_type(8)))  float  f32x8;
typedef unsigned short u16;

#define HID 256

__device__ __forceinline__ u16 f2bf(float f) {
    union { float f; uint32_t u; } v; v.f = f;
    uint32_t u = v.u;
    return (u16)((u + 0x7FFFu + ((u >> 16) & 1u)) >> 16);   // RNE
}

__device__ __forceinline__ float silu_f(float x) {
    return x / (1.0f + __expf(-x));
}

union BFrag { uint4 u[2]; bf16x16 v; };

// A fragment, 16x32 bf16 (ISA 7.12.2): lane<16 holds row M=lane, K halves
// {kk..kk+7, kk+16..kk+23}; lane>=16 holds {kk+8..kk+15, kk+24..kk+31}.
__device__ __forceinline__ bf16x16 load_frag_a(const u16* base, int row, int stride, int kk, int hi) {
    const u16* p = base + (size_t)row * stride + kk + hi * 8;
    BFrag c;
    c.u[0] = *(const uint4*)p;
    c.u[1] = *(const uint4*)(p + 16);
    return c.v;
}

// B fragment, 32x16 bf16, weights pre-transposed to [N][K]: lane holds column
// N=lane&15, K halves {kk+8*hi..+7, kk+16+8*hi..+7} (same K order as A).
__device__ __forceinline__ bf16x16 load_frag_b(const u16* wt, int n, int K, int kk, int hi) {
    const u16* p = wt + (size_t)n * K + kk + hi * 8;
    BFrag c;
    c.u[0] = *(const uint4*)p;
    c.u[1] = *(const uint4*)(p + 16);
    return c.v;
}

__device__ __forceinline__ f32x8 wmma_bf16(bf16x16 a, bf16x16 b, f32x8 c) {
    return __builtin_amdgcn_wmma_f32_16x16x32_bf16(false, a, false, b, (short)0, c, false, false);
}

// ---------------------------------------------------------------- utilities

__global__ void zero_f32_kernel(float4* __restrict__ p, int n4) {
    int i = blockIdx.x * 256 + threadIdx.x;
    if (i < n4) p[i] = make_float4(0.f, 0.f, 0.f, 0.f);
}

// dst[N][K] bf16 <- transpose(src[K][N] f32)
__global__ void convT_kernel(const float* __restrict__ src, u16* __restrict__ dst, int K, int N) {
    int idx = blockIdx.x * 256 + threadIdx.x;
    if (idx >= K * N) return;
    int n = idx / K, k = idx - n * K;
    dst[idx] = f2bf(src[(size_t)k * N + n]);
}

// ---------------------------------------------------------------- embed: h = h0 @ w_in + b_in

__global__ __launch_bounds__(256) void embed_kernel(
    const float* __restrict__ h0,      // [N,64]
    const u16*   __restrict__ wt_in,   // [256][64] bf16 (transposed)
    const float* __restrict__ b_in,    // [256]
    u16*         __restrict__ h_bf,    // [N,256] bf16 out
    int n_nodes)
{
    __shared__ u16 sA[32 * 72];        // 32 rows x 64 halves, padded stride
    __shared__ u16 sO[32 * 264];       // output staging
    int n0 = blockIdx.x * 32;
    int tid = threadIdx.x;
    int wave = tid >> 5, lane = tid & 31, hi = lane >> 4, ln = lane & 15;

    for (int i = 0; i < 8; ++i) {
        int idx = tid + i * 256;                  // 0..2047
        int r = idx >> 6, c = idx & 63;
        int node = min(n0 + r, n_nodes - 1);
        sA[r * 72 + c] = f2bf(h0[(size_t)node * 64 + c]);
    }
    __syncthreads();

    f32x8 acc[2][2] = {};
    const u16* sA1 = sA + 16 * 72;
#pragma unroll
    for (int kk = 0; kk < 64; kk += 32) {
        bf16x16 a0 = load_frag_a(sA,  ln, 72, kk, hi);
        bf16x16 a1 = load_frag_a(sA1, ln, 72, kk, hi);
#pragma unroll
        for (int t = 0; t < 2; ++t) {
            int n = (wave * 2 + t) * 16 + ln;
            bf16x16 b = load_frag_b(wt_in, n, 64, kk, hi);
            acc[0][t] = wmma_bf16(a0, b, acc[0][t]);
            acc[1][t] = wmma_bf16(a1, b, acc[1][t]);
        }
    }
#pragma unroll
    for (int t = 0; t < 2; ++t) {
        int n = (wave * 2 + t) * 16 + ln;
        float bb = b_in[n];
#pragma unroll
        for (int sl = 0; sl < 2; ++sl)
#pragma unroll
            for (int r = 0; r < 8; ++r) {
                int m = sl * 16 + r + hi * 8;
                sO[m * 264 + n] = f2bf(acc[sl][t][r] + bb);
            }
    }
    __syncthreads();
    // coalesced copy out: 32 rows x 32 uint4
    for (int i = 0; i < 4; ++i) {
        int idx = tid + i * 256;
        int r = idx >> 5, s = idx & 31;
        int node = n0 + r;
        if (node < n_nodes)
            *((uint4*)(h_bf + (size_t)node * HID) + s) = *((const uint4*)(sO + r * 264) + s);
    }
}

// ---------------------------------------------------------------- fused edge MLP + scatter-add
// 64 edges / block, 4 row-slabs per wave.

__global__ __launch_bounds__(256) void edge_kernel(
    const u16*     __restrict__ h_bf,   // [N,256] bf16
    const int64_t* __restrict__ rows,   // [E]
    const int64_t* __restrict__ cols,   // [E]
    const float*   __restrict__ cdiff,  // [E,3]
    const u16*     __restrict__ wt1,    // [256][512] bf16  (edge_w1 rows 0..511, transposed)
    const float*   __restrict__ w1rad,  // [256] f32        (edge_w1 row 512)
    const float*   __restrict__ b1,     // [256]
    const u16*     __restrict__ wt2,    // [256][256] bf16
    const float*   __restrict__ b2,     // [256]
    float*         __restrict__ agg,    // [N,256] f32 (atomic accum)
    int n_edges)
{
    __shared__ u16   sA[64 * 520];      // [h_row | h_col] bf16, padded stride (66.6 KB)
    __shared__ u16   sM[64 * 264];      // silu(m1) bf16, padded stride (33.8 KB)
    __shared__ float sRad[64];
    __shared__ int   sRow[64];

    int e0 = blockIdx.x * 64;
    int tid = threadIdx.x;
    int wave = tid >> 5, lane = tid & 31, hi = lane >> 4, ln = lane & 15;

    if (tid < 64) {
        int e = min(e0 + tid, n_edges - 1);
        sRow[tid] = (int)rows[e];
        float dx = cdiff[(size_t)e * 3 + 0];
        float dy = cdiff[(size_t)e * 3 + 1];
        float dz = cdiff[(size_t)e * 3 + 2];
        sRad[tid] = dx * dx + dy * dy + dz * dz;
    }
    // gather [h[row] | h[col]] -> LDS; 64 rows x 64 uint4 slots.
    // idx>>6 is wave-uniform -> rows[e]/cols[e] become scalar loads, and each
    // wave streams 512B contiguous from one h row.
    for (int i = 0; i < 16; ++i) {
        int idx = tid + i * 256;
        int r = idx >> 6, s = idx & 63;
        int e = min(e0 + r, n_edges - 1);
        int64_t src = (s < 32) ? rows[e] : cols[e];
        uint4 d = *((const uint4*)(h_bf + (size_t)src * HID) + (s & 31));
        *((uint4*)(sA + r * 520) + s) = d;
    }
    __syncthreads();

    // GEMM1: [64 x 512] @ [512 x 256]
    f32x8 acc[4][2] = {};
#pragma unroll 2
    for (int kk = 0; kk < 512; kk += 32) {
        bf16x16 a[4];
#pragma unroll
        for (int sl = 0; sl < 4; ++sl)
            a[sl] = load_frag_a(sA + sl * 16 * 520, ln, 520, kk, hi);
#pragma unroll
        for (int t = 0; t < 2; ++t) {
            int n = (wave * 2 + t) * 16 + ln;
            bf16x16 b = load_frag_b(wt1, n, 512, kk, hi);
#pragma unroll
            for (int sl = 0; sl < 4; ++sl)
                acc[sl][t] = wmma_bf16(a[sl], b, acc[sl][t]);
        }
    }
    // epilogue: + radial * w1[512] + b1, silu, -> LDS bf16
#pragma unroll
    for (int t = 0; t < 2; ++t) {
        int n = (wave * 2 + t) * 16 + ln;
        float wr = w1rad[n], bb = b1[n];
#pragma unroll
        for (int sl = 0; sl < 4; ++sl)
#pragma unroll
            for (int r = 0; r < 8; ++r) {
                int m = sl * 16 + r + hi * 8;
                float x = acc[sl][t][r] + sRad[m] * wr + bb;
                sM[m * 264 + n] = f2bf(silu_f(x));
            }
    }
    __syncthreads();

    // GEMM2: [64 x 256] @ [256 x 256]
    f32x8 acc2[4][2] = {};
#pragma unroll 2
    for (int kk = 0; kk < 256; kk += 32) {
        bf16x16 a[4];
#pragma unroll
        for (int sl = 0; sl < 4; ++sl)
            a[sl] = load_frag_a(sM + sl * 16 * 264, ln, 264, kk, hi);
#pragma unroll
        for (int t = 0; t < 2; ++t) {
            int n = (wave * 2 + t) * 16 + ln;
            bf16x16 b = load_frag_b(wt2, n, 256, kk, hi);
#pragma unroll
            for (int sl = 0; sl < 4; ++sl)
                acc2[sl][t] = wmma_bf16(a[sl], b, acc2[sl][t]);
        }
    }
    // epilogue: + b2, silu, scatter-add into agg[row].
    // Single path: out-of-range rows contribute exactly 0.0 (atomic no-op).
#pragma unroll
    for (int t = 0; t < 2; ++t) {
        int n = (wave * 2 + t) * 16 + ln;
        float bb = b2[n];
#pragma unroll
        for (int sl = 0; sl < 4; ++sl)
#pragma unroll
            for (int r = 0; r < 8; ++r) {
                int m = sl * 16 + r + hi * 8;
                float v = silu_f(acc2[sl][t][r] + bb);
                v = (e0 + m < n_edges) ? v : 0.0f;
                unsafeAtomicAdd(&agg[(size_t)sRow[m] * HID + n], v);
            }
    }
}

// ---------------------------------------------------------------- node MLP: h = silu([h|agg]@W1+b1)@W2+b2

__global__ __launch_bounds__(256) void node_kernel(
    u16*         __restrict__ h_bf,    // [N,256] bf16, updated in place
    const float* __restrict__ agg,     // [N,256] f32
    const u16*   __restrict__ wt1,     // [256][512] bf16
    const float* __restrict__ b1,
    const u16*   __restrict__ wt2,     // [256][256] bf16
    const float* __restrict__ b2,
    int n_nodes)
{
    __shared__ u16 sA[32 * 520];
    __shared__ u16 sM[32 * 264];

    int n0 = blockIdx.x * 32;
    int tid = threadIdx.x;
    int wave = tid >> 5, lane = tid & 31, hi = lane >> 4, ln = lane & 15;

    for (int i = 0; i < 8; ++i) {
        int idx = tid + i * 256;
        int r = idx >> 6, s = idx & 63;
        int node = min(n0 + r, n_nodes - 1);
        if (s < 32) {
            *((uint4*)(sA + r * 520) + s) =
                *((const uint4*)(h_bf + (size_t)node * HID) + s);
        } else {
            const float* ap = agg + (size_t)node * HID + (s - 32) * 8;
            u16* dp = sA + r * 520 + 256 + (s - 32) * 8;
#pragma unroll
            for (int j = 0; j < 8; ++j) dp[j] = f2bf(ap[j]);
        }
    }
    __syncthreads();

    f32x8 acc[2][2] = {};
    const u16* sA1 = sA + 16 * 520;
#pragma unroll 2
    for (int kk = 0; kk < 512; kk += 32) {
        bf16x16 a0 = load_frag_a(sA,  ln, 520, kk, hi);
        bf16x16 a1 = load_frag_a(sA1, ln, 520, kk, hi);
#pragma unroll
        for (int t = 0; t < 2; ++t) {
            int n = (wave * 2 + t) * 16 + ln;
            bf16x16 b = load_frag_b(wt1, n, 512, kk, hi);
            acc[0][t] = wmma_bf16(a0, b, acc[0][t]);
            acc[1][t] = wmma_bf16(a1, b, acc[1][t]);
        }
    }
#pragma unroll
    for (int t = 0; t < 2; ++t) {
        int n = (wave * 2 + t) * 16 + ln;
        float bb = b1[n];
#pragma unroll
        for (int sl = 0; sl < 2; ++sl)
#pragma unroll
            for (int r = 0; r < 8; ++r) {
                int m = sl * 16 + r + hi * 8;
                sM[m * 264 + n] = f2bf(silu_f(acc[sl][t][r] + bb));
            }
    }
    __syncthreads();   // all waves done with sA (GEMM1) and sM writes

    f32x8 acc2[2][2] = {};
    const u16* sM1 = sM + 16 * 264;
#pragma unroll 2
    for (int kk = 0; kk < 256; kk += 32) {
        bf16x16 a0 = load_frag_a(sM,  ln, 264, kk, hi);
        bf16x16 a1 = load_frag_a(sM1, ln, 264, kk, hi);
#pragma unroll
        for (int t = 0; t < 2; ++t) {
            int n = (wave * 2 + t) * 16 + ln;
            bf16x16 b = load_frag_b(wt2, n, 256, kk, hi);
            acc2[0][t] = wmma_bf16(a0, b, acc2[0][t]);
            acc2[1][t] = wmma_bf16(a1, b, acc2[1][t]);
        }
    }
    // stage final h (bf16) into sA (free after the barrier above), then copy
#pragma unroll
    for (int t = 0; t < 2; ++t) {
        int n = (wave * 2 + t) * 16 + ln;
        float bb = b2[n];
#pragma unroll
        for (int sl = 0; sl < 2; ++sl)
#pragma unroll
            for (int r = 0; r < 8; ++r) {
                int m = sl * 16 + r + hi * 8;
                sA[m * 264 + n] = f2bf(acc2[sl][t][r] + bb);
            }
    }
    __syncthreads();
    for (int i = 0; i < 4; ++i) {
        int idx = tid + i * 256;
        int r = idx >> 5, s = idx & 31;
        int node = n0 + r;
        if (node < n_nodes)
            *((uint4*)(h_bf + (size_t)node * HID) + s) = *((const uint4*)(sA + r * 264) + s);
    }
}

// ---------------------------------------------------------------- out: y = h @ w_out + b_out

__global__ __launch_bounds__(256) void out_kernel(
    const u16*   __restrict__ h_bf,    // [N,256] bf16 (row-major, used directly as A)
    const u16*   __restrict__ wt_out,  // [64][256] bf16 (transposed)
    const float* __restrict__ b_out,   // [64]
    float*       __restrict__ out,     // [N,64]
    int n_nodes)
{
    __shared__ float sO[32 * 68];      // 32 rows x 64 f32, padded stride
    int n0 = blockIdx.x * 32;
    int tid = threadIdx.x;
    int wave = tid >> 5, lane = tid & 31, hi = lane >> 4, ln = lane & 15;
    int tile = wave & 3, slab = wave >> 2;

    f32x8 acc = {};
    int arow = min(n0 + slab * 16 + ln, n_nodes - 1);
    int n = tile * 16 + ln;
#pragma unroll 2
    for (int kk = 0; kk < 256; kk += 32) {
        bf16x16 a = load_frag_a(h_bf, arow, 256, kk, hi);
        acc = wmma_bf16(a, load_frag_b(wt_out, n, 256, kk, hi), acc);
    }
    float bb = b_out[n];
#pragma unroll
    for (int r = 0; r < 8; ++r) {
        int m = slab * 16 + r + hi * 8;
        sO[m * 68 + n] = acc[r] + bb;
    }
    __syncthreads();
    // coalesced copy: 32 rows x 16 uint4 (64 f32)
    for (int i = 0; i < 2; ++i) {
        int idx = tid + i * 256;
        int r = idx >> 4, s = idx & 15;
        int node = n0 + r;
        if (node < n_nodes)
            *((uint4*)(out + (size_t)node * 64) + s) = *((const uint4*)(sO + r * 68) + s);
    }
}

// ---------------------------------------------------------------- host

extern "C" void kernel_launch(void* const* d_in, const int* in_sizes, int n_in,
                              void* d_out, int out_size, void* d_ws, size_t ws_size,
                              hipStream_t stream) {
    const float*   h0    = (const float*)d_in[0];
    const int64_t* ei    = (const int64_t*)d_in[1];
    const float*   cdiff = (const float*)d_in[2];
    const float*   w_in  = (const float*)d_in[3];
    const float*   b_in  = (const float*)d_in[4];
    const float*   w_out = (const float*)d_in[5];
    const float*   b_out = (const float*)d_in[6];
    const float*   e_w1  = (const float*)d_in[7];
    const float*   e_b1  = (const float*)d_in[8];
    const float*   e_w2  = (const float*)d_in[9];
    const float*   e_b2  = (const float*)d_in[10];
    const float*   n_w1  = (const float*)d_in[11];
    const float*   n_b1  = (const float*)d_in[12];
    const float*   n_w2  = (const float*)d_in[13];
    const float*   n_b2  = (const float*)d_in[14];

    const int n_nodes = in_sizes[0] / 64;
    const int n_edges = in_sizes[1] / 2;
    const int L = 4;

    char* ws = (char*)d_ws;
    size_t off = 0;
    auto salloc = [&](size_t bytes) -> void* {
        void* p = ws + off;
        off = (off + bytes + 255) & ~(size_t)255;
        return p;
    };
    u16*   h_bf   = (u16*)  salloc((size_t)n_nodes * HID * 2);
    float* agg    = (float*)salloc((size_t)n_nodes * HID * 4);
    u16*   wt_in  = (u16*)  salloc((size_t)256 * 64 * 2);
    u16*   wt_out = (u16*)  salloc((size_t)64 * 256 * 2);
    u16*   wt_e1  = (u16*)  salloc((size_t)L * 256 * 512 * 2);
    u16*   wt_e2  = (u16*)  salloc((size_t)L * 256 * 256 * 2);
    u16*   wt_n1  = (u16*)  salloc((size_t)L * 256 * 512 * 2);
    u16*   wt_n2  = (u16*)  salloc((size_t)L * 256 * 256 * 2);

    auto convT = [&](const float* src, u16* dst, int K, int N) {
        int total = K * N;
        convT_kernel<<<dim3((total + 255) / 256), dim3(256), 0, stream>>>(src, dst, K, N);
    };

    // one-time weight convert + transpose (bf16, [N][K])
    convT(w_in, wt_in, 64, 256);
    convT(w_out, wt_out, 256, 64);
    for (int l = 0; l < L; ++l) {
        convT(e_w1 + (size_t)l * 513 * 256, wt_e1 + (size_t)l * 256 * 512, 512, 256);
        convT(e_w2 + (size_t)l * 256 * 256, wt_e2 + (size_t)l * 256 * 256, 256, 256);
        convT(n_w1 + (size_t)l * 512 * 256, wt_n1 + (size_t)l * 256 * 512, 512, 256);
        convT(n_w2 + (size_t)l * 256 * 256, wt_n2 + (size_t)l * 256 * 256, 256, 256);
    }

    // embedding
    embed_kernel<<<dim3((n_nodes + 31) / 32), dim3(256), 0, stream>>>(
        h0, wt_in, b_in, h_bf, n_nodes);

    for (int l = 0; l < L; ++l) {
        int n4 = n_nodes * HID / 4;
        zero_f32_kernel<<<dim3((n4 + 255) / 256), dim3(256), 0, stream>>>((float4*)agg, n4);

        edge_kernel<<<dim3((n_edges + 63) / 64), dim3(256), 0, stream>>>(
            h_bf, ei, ei + n_edges, cdiff,
            wt_e1 + (size_t)l * 256 * 512,
            e_w1 + (size_t)l * 513 * 256 + (size_t)512 * 256,   // radial row of W1
            e_b1 + (size_t)l * 256,
            wt_e2 + (size_t)l * 256 * 256,
            e_b2 + (size_t)l * 256,
            agg, n_edges);

        node_kernel<<<dim3((n_nodes + 31) / 32), dim3(256), 0, stream>>>(
            h_bf, agg,
            wt_n1 + (size_t)l * 256 * 512, n_b1 + (size_t)l * 256,
            wt_n2 + (size_t)l * 256 * 256, n_b2 + (size_t)l * 256,
            n_nodes);
    }

    out_kernel<<<dim3((n_nodes + 31) / 32), dim3(256), 0, stream>>>(
        h_bf, wt_out, b_out, (float*)d_out, n_nodes);
}
